// HeadAttention_12575664242839
// MI455X (gfx1250) — matrix operand order repaired and checked by reading, not compiled, not run on hardware
//
#include <hip/hip_runtime.h>
#include <stdint.h>

typedef __attribute__((ext_vector_type(16))) __bf16 v16bf;
typedef __attribute__((ext_vector_type(8)))  __bf16 v8bf;
typedef __attribute__((ext_vector_type(4)))  __bf16 v4bf;
typedef __attribute__((ext_vector_type(8)))  float  v8f;
typedef __attribute__((ext_vector_type(4)))  float  fvec4;
typedef __attribute__((ext_vector_type(4)))  unsigned int u32x4;
typedef __attribute__((ext_vector_type(8)))  int    i32x8;
typedef __attribute__((ext_vector_type(4)))  int    i32x4;

#define EMB  2048
#define HEAD 128
#define SEQ  4096

#if __has_builtin(__builtin_amdgcn_tensor_load_to_lds) && __has_builtin(__builtin_amdgcn_s_wait_tensorcnt)
#define USE_TDM 1
#else
#define USE_TDM 0
#endif

// Scheduling directives: pin DS-read / WMMA interleave.
#if __has_builtin(__builtin_amdgcn_sched_group_barrier)
#define SGB(mask, n) __builtin_amdgcn_sched_group_barrier(mask, n, 0)
#else
#define SGB(mask, n)
#endif

static __device__ __forceinline__ v16bf cat16(v8bf lo, v8bf hi) {
  v16bf r;
#pragma unroll
  for (int i = 0; i < 8; ++i) { r[i] = lo[i]; r[i + 8] = hi[i]; }
  return r;
}

static __device__ __forceinline__ v8f wmma_bf16(v16bf a, v16bf b, v8f c) {
  return __builtin_amdgcn_wmma_f32_16x16x32_bf16(false, a, false, b, (short)0, c,
                                                 false, false);
}

#if USE_TDM
// 2-D TDM tile load: tile_d0 elems/row (bf16), tile_d1 rows, global row stride
// g_stride elems; LDS rows padded per D# pad fields (CDNA5 ISA).
static __device__ __forceinline__ void tdm_load_2d(
    unsigned lds_addr, unsigned long long gaddr,
    unsigned tensor_d0, unsigned tensor_d1, unsigned g_stride,
    unsigned tile_d0, unsigned tile_d1,
    unsigned pad_interval, unsigned pad_amount)
{
  u32x4 g0 = { 1u,                                   // count=1 valid user D#
               lds_addr,
               (unsigned)(gaddr & 0xFFFFFFFFu),
               (unsigned)((gaddr >> 32) & 0x01FFFFFFu) | (2u << 30) };  // type=2
  i32x8 g1 = { (int)((1u << 16) | (1u << 20) |        // data_size=2B, pad_enable
                     (pad_interval << 22) | (pad_amount << 25)),
               (int)(tensor_d0 << 16),                // tensor_dim0[15:0]
               (int)(tensor_d1 << 16),                // dim0 hi=0 | tensor_dim1 lo16
               (int)(tile_d0 << 16),                  // dim1 hi=0 | tile_dim0
               (int)tile_d1,                          // tile_dim1 | tile_dim2=0
               (int)g_stride,                         // tensor_dim0_stride lo32
               0, 0 };
  i32x4 gz = {0, 0, 0, 0};
#if defined(__clang_major__) && __clang_major__ >= 23
  i32x8 gz8 = {0, 0, 0, 0, 0, 0, 0, 0};
  __builtin_amdgcn_tensor_load_to_lds(g0, g1, gz, gz, gz8, 0);
#else
  __builtin_amdgcn_tensor_load_to_lds(g0, g1, gz, gz, 0);
#endif
}
#endif

// ---------------------------------------------------------------------------
// Kernel 0: one-time weight transpose+convert: w[E][H] f32 -> wt[3][H][E] bf16
// ---------------------------------------------------------------------------
__global__ __launch_bounds__(256) void wtrans_kernel(
    const float* __restrict__ w0, const float* __restrict__ w1,
    const float* __restrict__ w2, __bf16* __restrict__ wt)
{
  const int w = blockIdx.y;
  const float* wp = (w == 0) ? w0 : (w == 1) ? w1 : w2;
  int idx = blockIdx.x * 256 + threadIdx.x;     // 0 .. EMB*HEAD-1
  int k = idx >> 7;                             // / HEAD
  int n = idx & (HEAD - 1);
  wt[(size_t)w * HEAD * EMB + (size_t)n * EMB + k] = (__bf16)wp[idx];
}

// ---------------------------------------------------------------------------
// Kernel 1: fused QKV projection.  [16384 x 2048] @ [2048 x 128] x3.
// LDS ping-pong: TDM weight fills + x conversion for step i+1 overlap with
// 24 WMMAs of step i.  Depth-2 register pipeline on B fragments.
// ---------------------------------------------------------------------------
__global__ __launch_bounds__(256) void qkv_proj_kernel(
    const float* __restrict__ x, const __bf16* __restrict__ wt,
    const float* __restrict__ b0p, __bf16* __restrict__ o0, float s0,
    const float* __restrict__ b1p, __bf16* __restrict__ o1, float s1,
    const float* __restrict__ b2p, __bf16* __restrict__ o2, float s2)
{
  __shared__ __bf16 xA[2][128 * 40];      // x tile [row][k], stride 40
  __shared__ __bf16 wTl[2][3][128 * 40];  // weight tiles [n][k], stride 40

  const int tid  = threadIdx.x;
  const int lane = tid & 31;
  const int wave = tid >> 5;
  const int half = lane >> 4;
  const int l16  = lane & 15;
  const int mBase = blockIdx.x * 128;

  v8f acc[3][8];
#pragma unroll
  for (int w = 0; w < 3; ++w)
#pragma unroll
    for (int nt = 0; nt < 8; ++nt)
#pragma unroll
      for (int j = 0; j < 8; ++j) acc[w][nt][j] = 0.0f;

  // Stage K-step (weights via TDM, x converted f32->bf16 by VALU)
  auto stage = [&](int e, int buf) {
#if USE_TDM
    if (wave == 0) {
      // 128 rows x 32 bf16, row stride EMB; 16 DW data + 4 DW pad per row
#pragma unroll
      for (int w = 0; w < 3; ++w)
        tdm_load_2d((unsigned)(uintptr_t)(&wTl[buf][w][0]),
                    (unsigned long long)(uintptr_t)(wt + (size_t)w * HEAD * EMB + e),
                    EMB, HEAD, EMB, 32, 128, 3u, 3u);
    }
#else
#pragma unroll
    for (int i = 0; i < 6; ++i) {
      int c = tid + i * 256;              // 0..1535 chunks of 8 bf16
      int w = c / 512;
      int r = c & 511;
      int n = r >> 2, k0 = (r & 3) << 3;
      *(v8bf*)(&wTl[buf][w][n * 40 + k0]) =
          *(const v8bf*)(wt + (size_t)w * HEAD * EMB + (size_t)n * EMB + e + k0);
    }
#endif
#pragma unroll
    for (int i = 0; i < 4; ++i) {
      int c = tid + i * 256;              // 0..1023 16B chunks
      int r = c >> 3, f0 = (c & 7) << 2;
      fvec4 v = *(const fvec4*)(x + (size_t)(mBase + r) * EMB + e + f0);
      v4bf b4;
#pragma unroll
      for (int k = 0; k < 4; ++k) b4[k] = (__bf16)v[k];
      *(v4bf*)(&xA[buf][r * 40 + f0]) = b4;
    }
  };

  stage(0, 0);                            // prologue fill of buffer 0

  for (int it = 0; it < EMB / 32; ++it) {
    const int e   = it * 32;
    const int cur = it & 1;
#if USE_TDM
    if (wave == 0) __builtin_amdgcn_s_wait_tensorcnt(0);  // buf[cur] TDM done
#endif
    __syncthreads();                      // buf[cur] ready; buf[cur^1] free
    if (e + 32 < EMB) stage(e + 32, cur ^ 1);

    // A fragment: lane<16 -> row=l16, K in {kb..kb+7, kb+16..kb+23}, kb=half*8
    const int arow = wave * 16 + l16;
    const int akb  = half * 8;
    v16bf afrag = cat16(*(const v8bf*)(&xA[cur][arow * 40 + akb]),
                        *(const v8bf*)(&xA[cur][arow * 40 + akb + 16]));

    // 24 WMMAs, depth-2 register pipeline (two live B-fragment sets)
    const int bkb = half * 8;
    auto loadB = [&](int idx) -> v16bf {
      const __bf16* bp = &wTl[cur][idx >> 3][((idx & 7) * 16 + l16) * 40 + bkb];
      return cat16(*(const v8bf*)bp, *(const v8bf*)(bp + 16));
    };
    v16bf b0 = loadB(0);
    v16bf b1 = loadB(1);
#pragma unroll
    for (int idx = 0; idx < 24; ++idx) {
      acc[idx >> 3][idx & 7] =
          wmma_bf16(afrag, (idx & 1) ? b1 : b0, acc[idx >> 3][idx & 7]);
      if (idx + 2 < 24) {
        if (idx & 1) b1 = loadB(idx + 2); else b0 = loadB(idx + 2);
      }
    }
    // Pin issue order: A + B0 + B1 reads, then (wmma, 2 reads) x22, 2 wmmas
    SGB(0x100, 6);
#pragma unroll
    for (int g = 0; g < 22; ++g) { SGB(0x008, 1); SGB(0x100, 2); }
    SGB(0x008, 2);
  }

  const float* bArr[3] = {b0p, b1p, b2p};
  __bf16*      oArr[3] = {o0, o1, o2};
  const float  sArr[3] = {s0, s1, s2};
#pragma unroll
  for (int w = 0; w < 3; ++w) {
#pragma unroll
    for (int nt = 0; nt < 8; ++nt) {
      int n = nt * 16 + l16;
      float bv = bArr[w][n];
#pragma unroll
      for (int j = 0; j < 8; ++j) {
        int row = mBase + wave * 16 + j + half * 8;   // C layout: VGPR j -> M=j(+8)
        oArr[w][(size_t)row * HEAD + n] = (__bf16)((acc[w][nt][j] + bv) * sArr[w]);
      }
    }
  }
}

// ---------------------------------------------------------------------------
// Kernel 2: causal flash attention.  Block = (batch, 128-query tile), 8 waves.
// LDS ping-pong: TDM K-tile + VALU V-transpose for tile i+1 overlap with the
// 32 WMMAs + softmax of tile i.  Depth-2 register pipelines.
// ---------------------------------------------------------------------------
__global__ __launch_bounds__(256) void attn_kernel(
    const __bf16* __restrict__ Qg, const __bf16* __restrict__ Kg,
    const __bf16* __restrict__ Vg, float* __restrict__ out)
{
  __shared__ __bf16 Kt[2][64 * 136];    // K tile [key][h], stride 136
  __shared__ __bf16 Vt[2][128 * 72];    // V tile transposed [h][key], stride 72
  __shared__ __bf16 Pb[8][16 * 72];     // per-wave P re-layout scratch

  const int tid  = threadIdx.x;
  const int lane = tid & 31;
  const int wave = tid >> 5;
  const int half = lane >> 4;
  const int l16  = lane & 15;

  const int qt    = blockIdx.x;
  const int batch = blockIdx.y;
  const int qBase = qt * 128;
  const size_t base = (size_t)batch * SEQ * HEAD;
  const __bf16* Q = Qg + base;
  const __bf16* K = Kg + base;
  const __bf16* V = Vg + base;

  // Q fragments (K-dim = HEAD = 128 -> 4 frags), 1/sqrt(H) pre-folded into Q
  v16bf qf[4];
  {
    int qrow = qBase + wave * 16 + l16;
#pragma unroll
    for (int f = 0; f < 4; ++f) {
      int kb = f * 32 + half * 8;
      qf[f] = cat16(*(const v8bf*)(Q + (size_t)qrow * HEAD + kb),
                    *(const v8bf*)(Q + (size_t)qrow * HEAD + kb + 16));
    }
  }

  v8f  o[8];
  float m[8], l[8];
#pragma unroll
  for (int i = 0; i < 8; ++i) {
#pragma unroll
    for (int j = 0; j < 8; ++j) o[i][j] = 0.0f;
    m[i] = -3.0e38f;
    l[i] = 0.0f;
  }

  // Stage one key tile (K via TDM with padded rows; V transposed by VALU)
  auto stageKV = [&](int kBase, int buf) {
#if USE_TDM
    if (wave == 0) {
      // 64 rows x 128 bf16, row stride HEAD; 64 DW data + 4 DW pad per row
      tdm_load_2d((unsigned)(uintptr_t)(&Kt[buf][0]),
                  (unsigned long long)(uintptr_t)(K + (size_t)kBase * HEAD),
                  HEAD, SEQ, HEAD, 128, 64, 5u, 3u);
    }
#else
#pragma unroll
    for (int i = 0; i < 4; ++i) {
      int c = tid + i * 256;
      int key = c >> 4, h0 = (c & 15) << 3;
      *(v8bf*)(&Kt[buf][key * 136 + h0]) =
          *(const v8bf*)(K + (size_t)(kBase + key) * HEAD + h0);
    }
#endif
#pragma unroll
    for (int i = 0; i < 4; ++i) {
      int c = tid + i * 256;
      int key = c >> 4, h0 = (c & 15) << 3;
      v8bf v = *(const v8bf*)(V + (size_t)(kBase + key) * HEAD + h0);
#pragma unroll
      for (int e = 0; e < 8; ++e) Vt[buf][(h0 + e) * 72 + key] = v[e];
    }
  };

  const int nkt = (qBase + 128) / 64;   // causal: keys <= qBase+127
  stageKV(0, 0);                        // prologue fill of buffer 0

  for (int kt = 0; kt < nkt; ++kt) {
    const int kBase = kt * 64;
    const int cur   = kt & 1;
#if USE_TDM
    if (wave == 0) __builtin_amdgcn_s_wait_tensorcnt(0);  // buf[cur] TDM done
#endif
    __syncthreads();                    // buf[cur] ready; buf[cur^1] free
    if (kt + 1 < nkt) stageKV(kBase + 64, cur ^ 1);
    if (kt + 2 < nkt)                   // warm L2 for the tile after next
      __builtin_prefetch(K + (size_t)(kBase + 128) * HEAD + tid * 32, 0, 0);

    // S = Q * K^T : 16 WMMAs, depth-2 register pipeline
    v8f sc[4];
#pragma unroll
    for (int nt = 0; nt < 4; ++nt)
#pragma unroll
      for (int j = 0; j < 8; ++j) sc[nt][j] = 0.0f;
    {
      auto loadK = [&](int idx) -> v16bf {
        const __bf16* bp =
            &Kt[cur][((idx >> 2) * 16 + l16) * 136 + (idx & 3) * 32 + half * 8];
        return cat16(*(const v8bf*)bp, *(const v8bf*)(bp + 16));
      };
      v16bf b0 = loadK(0);
      v16bf b1 = loadK(1);
#pragma unroll
      for (int idx = 0; idx < 16; ++idx) {
        sc[idx >> 2] = wmma_bf16(qf[idx & 3], (idx & 1) ? b1 : b0, sc[idx >> 2]);
        if (idx + 2 < 16) {
          if (idx & 1) b1 = loadK(idx + 2); else b0 = loadK(idx + 2);
        }
      }
      SGB(0x100, 4);
#pragma unroll
      for (int g = 0; g < 14; ++g) { SGB(0x008, 1); SGB(0x100, 2); }
      SGB(0x008, 2);
    }

    // Causal mask (diagonal-straddling tiles only; wave-uniform branch)
    if (kBase + 63 > qBase + wave * 16) {
#pragma unroll
      for (int nt = 0; nt < 4; ++nt) {
        int key = kBase + nt * 16 + l16;
#pragma unroll
        for (int j = 0; j < 8; ++j) {
          int qr = qBase + wave * 16 + j + half * 8;
          if (key > qr) sc[nt][j] = -3.0e38f;
        }
      }
    }

    // Online softmax (row = (j, lane-half); reduce across 16 lanes)
#pragma unroll
    for (int j = 0; j < 8; ++j) {
      float rm = fmaxf(fmaxf(sc[0][j], sc[1][j]), fmaxf(sc[2][j], sc[3][j]));
      rm = fmaxf(rm, __shfl_xor(rm, 1, 32));
      rm = fmaxf(rm, __shfl_xor(rm, 2, 32));
      rm = fmaxf(rm, __shfl_xor(rm, 4, 32));
      rm = fmaxf(rm, __shfl_xor(rm, 8, 32));
      float mn = fmaxf(m[j], rm);
      float alpha = __expf(m[j] - mn);
      m[j] = mn;
      float rs = 0.0f;
#pragma unroll
      for (int nt = 0; nt < 4; ++nt) {
        float p = __expf(sc[nt][j] - mn);
        sc[nt][j] = p;
        rs += p;
      }
      rs += __shfl_xor(rs, 1, 32);
      rs += __shfl_xor(rs, 2, 32);
      rs += __shfl_xor(rs, 4, 32);
      rs += __shfl_xor(rs, 8, 32);
      l[j] = l[j] * alpha + rs;
#pragma unroll
      for (int nh = 0; nh < 8; ++nh) o[nh][j] *= alpha;
    }

    // Re-layout P (C layout -> A fragments) through wave-private LDS
    __bf16* P = Pb[wave];
#pragma unroll
    for (int nt = 0; nt < 4; ++nt)
#pragma unroll
      for (int j = 0; j < 8; ++j)
        P[(j + half * 8) * 72 + nt * 16 + l16] = (__bf16)sc[nt][j];

    v16bf pf[2];
#pragma unroll
    for (int f = 0; f < 2; ++f) {
      int kb = f * 32 + half * 8;
      pf[f] = cat16(*(const v8bf*)(&P[l16 * 72 + kb]),
                    *(const v8bf*)(&P[l16 * 72 + kb + 16]));
    }

    // O += P * V : 16 WMMAs, depth-2 register pipeline
    {
      auto loadV = [&](int idx) -> v16bf {
        const __bf16* bp =
            &Vt[cur][((idx >> 1) * 16 + l16) * 72 + (idx & 1) * 32 + half * 8];
        return cat16(*(const v8bf*)bp, *(const v8bf*)(bp + 16));
      };
      v16bf b0 = loadV(0);
      v16bf b1 = loadV(1);
#pragma unroll
      for (int idx = 0; idx < 16; ++idx) {
        o[idx >> 1] = wmma_bf16(pf[idx & 1], (idx & 1) ? b1 : b0, o[idx >> 1]);
        if (idx + 2 < 16) {
          if (idx & 1) b1 = loadV(idx + 2); else b0 = loadV(idx + 2);
        }
      }
      // P stores first, then P-frags + 2 V-frags, then (wmma, 2 reads) x14
      SGB(0x200, 32);
      SGB(0x100, 8);
#pragma unroll
      for (int g = 0; g < 14; ++g) { SGB(0x008, 1); SGB(0x100, 2); }
      SGB(0x008, 2);
    }
  }

  // Normalize and write fp32 output
#pragma unroll
  for (int j = 0; j < 8; ++j) {
    float inv = 1.0f / l[j];
    int row = qBase + wave * 16 + j + half * 8;
#pragma unroll
    for (int nh = 0; nh < 8; ++nh)
      out[base + (size_t)row * HEAD + nh * 16 + l16] = o[nh][j] * inv;
  }
}

// ---------------------------------------------------------------------------
extern "C" void kernel_launch(void* const* d_in, const int* in_sizes, int n_in,
                              void* d_out, int out_size, void* d_ws, size_t ws_size,
                              hipStream_t stream) {
  (void)in_sizes; (void)n_in; (void)out_size; (void)ws_size;
  const float* x  = (const float*)d_in[0];
  const float* wK = (const float*)d_in[1];
  const float* bK = (const float*)d_in[2];
  const float* wQ = (const float*)d_in[3];
  const float* bQ = (const float*)d_in[4];
  const float* wV = (const float*)d_in[5];
  const float* bV = (const float*)d_in[6];
  float* out = (float*)d_out;

  const size_t rows = (size_t)4 * SEQ;            // 16384
  __bf16* q_ws = (__bf16*)d_ws;
  __bf16* k_ws = q_ws + rows * HEAD;
  __bf16* v_ws = k_ws + rows * HEAD;
  __bf16* wt   = v_ws + rows * HEAD;              // [3][HEAD][EMB] bf16

  const float qscale = 0.08838834764831845f;      // 1/sqrt(128)

  // One-time weight transpose: w[E][H] f32 -> wt[w][H][E] bf16  (order Q,K,V)
  wtrans_kernel<<<dim3(EMB * HEAD / 256, 3), dim3(256), 0, stream>>>(wQ, wK, wV, wt);

  // QKV projection (x tile reused across 3 weights; weights via TDM)
  qkv_proj_kernel<<<dim3(128), dim3(256), 0, stream>>>(
      x, wt,
      bQ, q_ws, qscale,
      bK, k_ws, 1.0f,
      bV, v_ws, 1.0f);

  // Causal flash attention (K tiles via TDM)
  attn_kernel<<<dim3(SEQ / 128, 4), dim3(256), 0, stream>>>(q_ws, k_ws, v_ws, out);
}
